// RecurrentStateEmbedder_66597762892289
// MI455X (gfx1250) — matrix operand order repaired and checked by reading, not compiled, not run on hardware
//
#include <hip/hip_runtime.h>

// ---------------------------------------------------------------------------
// LSTM over [B=64, S=512, D=256, H=512] for MI455X (gfx1250, wave32, WMMA).
//
// Strategy (see analysis): single persistent kernel runs the full 512-step
// scan. 32 blocks x 128 threads (4 waves). Block j owns hidden cols
// [16j,16j+16); wave w owns batch rows [16w,16w+16). Each wave holds four
// 16x16 f32 accumulators (i,f,g,o gate tiles) and the cell state c stays in
// registers for the entire scan. W_ih / W_hh live in LDS as pre-swizzled
// bf16 WMMA B-fragments; x_proj is never materialized (recomputed per step
// with WMMA against the LDS-resident W_ih). h ping-pongs through a small
// bf16 buffer (L2-resident). Global sync per step via atomic counter.
// ---------------------------------------------------------------------------

#define B_  64
#define S_  512
#define D_  256
#define H_  512
#define G4_ (4 * H_)   // 2048

typedef __attribute__((ext_vector_type(16))) __bf16 v16bf;
typedef __attribute__((ext_vector_type(8)))  float  v8f;

union Frag {
    unsigned u[8];
    uint4    q[2];
    v16bf    v;
};

// round-to-nearest-even f32 -> bf16 (as u16)
__device__ __forceinline__ unsigned short f2bf(float f) {
    unsigned u = __float_as_uint(f);
    u = (u + 0x7FFFu + ((u >> 16) & 1u)) >> 16;
    return (unsigned short)u;
}
__device__ __forceinline__ unsigned packbf(float lo, float hi) {
    return (unsigned)f2bf(lo) | ((unsigned)f2bf(hi) << 16);
}
__device__ __forceinline__ float sigmoidf_(float x) {
    return 1.0f / (1.0f + __expf(-x));
}
__device__ __forceinline__ float tanhf_(float x) {
    // robust fast tanh: 1 - 2/(e^{2x}+1); exact limits at +-inf
    float e = __expf(2.0f * x);
    return 1.0f - 2.0f / (e + 1.0f);
}

// ---------------------------------------------------------------------------
// Workspace layout (bytes)
// ---------------------------------------------------------------------------
#define OFF_WIHP  0u                               // (D/2)*4H dwords = 1 MB
#define OFF_WHHP  (OFF_WIHP + (D_/2)*G4_*4u)       // (H/2)*4H dwords = 2 MB
#define OFF_BIAS  (OFF_WHHP + (H_/2)*G4_*4u)       // 4H floats = 8 KB
#define OFF_XBF   (OFF_BIAS + G4_*4u)              // B*S*D bf16 = 16 MB
#define OFF_HPING (OFF_XBF + (unsigned)(B_*S_*(D_/2))*4u) // 2 * B*H bf16
#define OFF_BAR   (OFF_HPING + 2u*B_*H_*2u)

// ---------------------------------------------------------------------------
// Kernel 1: pack weights to pair-major bf16, sum biases, seed h ping, zero bar
//   WhhP[kp*4H + g] = bf16pair(Whh[g*H + 2kp], Whh[g*H + 2kp+1])
//   WihP[kp*4H + g] = bf16pair(Wih[g*D + 2kp], Wih[g*D + 2kp+1])
// ---------------------------------------------------------------------------
__global__ __launch_bounds__(256) void lstm_pack_kernel(
    const float* __restrict__ Wih, const float* __restrict__ Whh,
    const float* __restrict__ bih, const float* __restrict__ bhh,
    const float* __restrict__ h0,
    unsigned* __restrict__ WihP, unsigned* __restrict__ WhhP,
    float* __restrict__ biasSum, unsigned short* __restrict__ hping0,
    unsigned long long* __restrict__ bar)
{
    int i = blockIdx.x * 256 + threadIdx.x;
    if (i < (H_ / 2) * G4_) {
        int kp = i / G4_, g = i % G4_;
        WhhP[i] = packbf(Whh[g * H_ + 2 * kp], Whh[g * H_ + 2 * kp + 1]);
    }
    if (i < (D_ / 2) * G4_) {
        int kp = i / G4_, g = i % G4_;
        WihP[i] = packbf(Wih[g * D_ + 2 * kp], Wih[g * D_ + 2 * kp + 1]);
    }
    if (i < G4_)      biasSum[i] = bih[i] + bhh[i];
    if (i < B_ * H_)  hping0[i]  = f2bf(h0[i]);
    if (i == 0)       *bar = 0ull;
}

// ---------------------------------------------------------------------------
// Kernel 2: X fp32 -> bf16 (pair-packed dwords, same [B,S,D] layout)
// ---------------------------------------------------------------------------
__global__ __launch_bounds__(256) void lstm_convx_kernel(
    const float* __restrict__ X, unsigned* __restrict__ Xbf)
{
    int i = blockIdx.x * 256 + threadIdx.x;            // dword index
    if (i < B_ * S_ * (D_ / 2)) {
        const float2 x = ((const float2*)X)[i];
        Xbf[i] = packbf(x.x, x.y);
    }
}

// ---------------------------------------------------------------------------
// Kernel 3: persistent LSTM scan
// ---------------------------------------------------------------------------
__global__ __launch_bounds__(128) void lstm_scan_kernel(
    const unsigned* __restrict__ Xbf,
    const unsigned* __restrict__ WihP,
    const unsigned* __restrict__ WhhP,
    const float* __restrict__ biasSum,
    const float* __restrict__ c0,
    unsigned short* __restrict__ hping,   // 2 * [B][H] bf16
    float* __restrict__ out,              // embeddings | hT | cT
    unsigned long long* __restrict__ bar)
{
    // LDS-resident B fragments, lane-major for conflict-free ds_load_b32:
    //   index = ((gate*KC + kc)*8 + v)*32 + lane
    __shared__ unsigned ldsWih[4 * 8  * 8 * 32];   // 32 KB
    __shared__ unsigned ldsWhh[4 * 16 * 8 * 32];   // 64 KB

    const int j    = blockIdx.x;            // hidden-column tile 0..31
    const int wave = threadIdx.x >> 5;      // row tile 0..3
    const int lane = threadIdx.x & 31;

    // ---- one-time LDS fill: B-fragment swizzle of packed weights ----------
    // B frag as packed here (contiguous k): VGPR v, lane L: N=L%16, Kpair = 16*kc + v + (L<16?0:8)
    for (int e = threadIdx.x; e < 4 * 8 * 32 * 8; e += 128) {
        int v = e & 7, L = (e >> 3) & 31, kc = (e >> 8) & 7, g = e >> 11;
        int kp  = 16 * kc + v + ((L < 16) ? 0 : 8);
        int col = 512 * g + 16 * j + (L & 15);
        ldsWih[((g * 8 + kc) * 8 + v) * 32 + L] = WihP[kp * G4_ + col];
    }
    for (int e = threadIdx.x; e < 4 * 16 * 32 * 8; e += 128) {
        int v = e & 7, L = (e >> 3) & 31, kc = (e >> 8) & 15, g = e >> 12;
        int kp  = 16 * kc + v + ((L < 16) ? 0 : 8);
        int col = 512 * g + 16 * j + (L & 15);
        ldsWhh[((g * 16 + kc) * 8 + v) * 32 + L] = WhhP[kp * G4_ + col];
    }
    __syncthreads();

    const int M      = lane & 15;
    const int hcol   = 16 * j + M;                 // this lane's hidden column
    const int rowOff = (lane < 16) ? 0 : 8;        // C/D row offset per ISA layout
    const int b0     = 16 * wave;                  // batch rows b0..b0+15
    const int arow   = b0 + M;                     // A-frag row (lane L -> M=L%16)
    const int aoffA  = (lane < 16) ? 0 : 4;        // A-frag offset in dword pairs (K-pair base 0 / 8)

    float biasv[4];
#pragma unroll
    for (int g = 0; g < 4; ++g) biasv[g] = biasSum[512 * g + hcol];

    float c[8];
#pragma unroll
    for (int r = 0; r < 8; ++r) c[r] = c0[(size_t)(b0 + r + rowOff) * H_ + hcol];

    float* const outHT = out + (size_t)B_ * S_ * H_;
    float* const outCT = outHT + B_ * H_;

    unsigned long long tgt = 0;

    for (int t = 0; t < S_; ++t) {
        v8f acc[4];
#pragma unroll
        for (int g = 0; g < 4; ++g)
#pragma unroll
            for (int r = 0; r < 8; ++r) acc[g][r] = biasv[g];

        // ---- input projection: acc += X[b, t, :] @ Wih^T (K = D = 256) ----
        {
            const uint4* xrow =
                (const uint4*)(Xbf + (size_t)(arow * S_ + t) * (D_ / 2));
#pragma unroll
            for (int kc = 0; kc < 8; ++kc) {
                Frag a;
                a.q[0] = xrow[4 * kc + (aoffA >> 1)];
                a.q[1] = xrow[4 * kc + 1 + (aoffA >> 1)];
#pragma unroll
                for (int g = 0; g < 4; ++g) {
                    Frag b;
#pragma unroll
                    for (int v = 0; v < 8; ++v)
                        b.u[v] = ldsWih[(((g * 8 + kc) * 8 + v) << 5) + lane];
                    acc[g] = __builtin_amdgcn_wmma_f32_16x16x32_bf16(
                        false, a.v, false, b.v, (short)0, acc[g], false, false);
                }
            }
        }

        // ---- recurrent: acc += h @ Whh^T (K = H = 512) --------------------
        {
            const unsigned* hrowd = (const unsigned*)hping +
                (size_t)(t & 1) * (B_ * H_ / 2) + (size_t)arow * (H_ / 2);
            const uint4* hrow = (const uint4*)hrowd;
#pragma unroll 4
            for (int kc = 0; kc < 16; ++kc) {
                Frag a;
                a.q[0] = hrow[4 * kc + (aoffA >> 1)];
                a.q[1] = hrow[4 * kc + 1 + (aoffA >> 1)];
#pragma unroll
                for (int g = 0; g < 4; ++g) {
                    Frag b;
#pragma unroll
                    for (int v = 0; v < 8; ++v)
                        b.u[v] = ldsWhh[(((g * 16 + kc) * 8 + v) << 5) + lane];
                    acc[g] = __builtin_amdgcn_wmma_f32_16x16x32_bf16(
                        false, a.v, false, b.v, (short)0, acc[g], false, false);
                }
            }
        }

        // ---- LSTM cell (pure per-lane math; c lives in registers) ---------
        unsigned short* hnext =
            hping + (size_t)((t + 1) & 1) * (B_ * H_);
        float hn[8];
#pragma unroll
        for (int r = 0; r < 8; ++r) {
            float iv = sigmoidf_(acc[0][r]);
            float fv = sigmoidf_(acc[1][r]);
            float gv = tanhf_(acc[2][r]);
            float ov = sigmoidf_(acc[3][r]);
            c[r]  = fv * c[r] + iv * gv;
            hn[r] = ov * tanhf_(c[r]);
            const int b = b0 + r + rowOff;
            out[((size_t)b * S_ + t) * H_ + hcol] = hn[r];         // embeddings
            hnext[(size_t)b * H_ + hcol] = f2bf(hn[r]);            // next-step A
        }
        if (t == S_ - 1) {
#pragma unroll
            for (int r = 0; r < 8; ++r) {
                const int b = b0 + r + rowOff;
                outHT[(size_t)b * H_ + hcol] = hn[r];
                outCT[(size_t)b * H_ + hcol] = c[r];
            }
        }

        // ---- device-wide step barrier -------------------------------------
        __threadfence();
        __syncthreads();
        if (threadIdx.x == 0)
            __hip_atomic_fetch_add(bar, 1ull, __ATOMIC_RELEASE,
                                   __HIP_MEMORY_SCOPE_AGENT);
        tgt += (unsigned long long)gridDim.x;
        if (threadIdx.x == 0) {
            while (__hip_atomic_load(bar, __ATOMIC_ACQUIRE,
                                     __HIP_MEMORY_SCOPE_AGENT) < tgt)
                __builtin_amdgcn_s_sleep(8);
        }
        __syncthreads();
        __threadfence();
    }
}

// ---------------------------------------------------------------------------
extern "C" void kernel_launch(void* const* d_in, const int* in_sizes, int n_in,
                              void* d_out, int out_size, void* d_ws, size_t ws_size,
                              hipStream_t stream) {
    const float* X   = (const float*)d_in[0];   // [B,S,D]
    const float* Wih = (const float*)d_in[1];   // [4H,D]
    const float* Whh = (const float*)d_in[2];   // [4H,H]
    const float* bih = (const float*)d_in[3];   // [4H]
    const float* bhh = (const float*)d_in[4];   // [4H]
    const float* h0  = (const float*)d_in[5];   // [B,H]
    const float* c0  = (const float*)d_in[6];   // [B,H]

    char* ws = (char*)d_ws;
    unsigned*           WihP    = (unsigned*)(ws + OFF_WIHP);
    unsigned*           WhhP    = (unsigned*)(ws + OFF_WHHP);
    float*              biasSum = (float*)(ws + OFF_BIAS);
    unsigned*           Xbf     = (unsigned*)(ws + OFF_XBF);
    unsigned short*     hping   = (unsigned short*)(ws + OFF_HPING);
    unsigned long long* bar     = (unsigned long long*)(ws + OFF_BAR);

    // pack weights / biases / h0 / barrier  (covers (H/2)*4H = 524288 items)
    lstm_pack_kernel<<<(H_ / 2) * G4_ / 256, 256, 0, stream>>>(
        Wih, Whh, bih, bhh, h0, WihP, WhhP, biasSum, hping, bar);

    // X -> bf16 pairs
    lstm_convx_kernel<<<B_ * S_ * (D_ / 2) / 256, 256, 0, stream>>>(X, Xbf);

    // persistent scan: 32 blocks (one per hidden-column tile) x 4 waves
    lstm_scan_kernel<<<H_ / 16, 128, 0, stream>>>(
        Xbf, WihP, WhhP, biasSum, c0, hping, (float*)d_out, bar);
}
